// CrissCrossAttention_30717606101404
// MI455X (gfx1250) — matrix operand order repaired and checked by reading, not compiled, not run on hardware
//
#include <hip/hip_runtime.h>
#include <hip/hip_bf16.h>

typedef _Float16 h16;
typedef __attribute__((ext_vector_type(16))) _Float16 v16h;
typedef __attribute__((ext_vector_type(8)))  float    v8f;
typedef __attribute__((ext_vector_type(4)))  unsigned int u32x4;
typedef __attribute__((ext_vector_type(8)))  int i32x8;
typedef __attribute__((ext_vector_type(4)))  int i32x4;

union Frag {
    v16h v;
    uint4 q[2];
    uint2 d[4];
    unsigned short s[16];
    h16 h[16];
};

#define NB    4
#define CIN_  256
#define C2_   512
#define CQK_  32
#define H_    128
#define W_    128
#define NP    16384   // H_*W_

#if __has_builtin(__builtin_amdgcn_tensor_load_to_lds) && __has_builtin(__builtin_amdgcn_s_wait_tensorcnt)
#define HAS_TDM 1
#else
#define HAS_TDM 0
#endif

#if HAS_TDM
static constexpr int NBUF = 2;   // double-buffered LDS for TDM overlap
// 2-D tensor tile -> LDS via Tensor Data Mover (D# per CDNA5 ISA 8.3/8.4).
// tile_d0 contiguous f16 elements per line, tile_d1 lines, line pitch = stride elems.
// TDM pad: after every 16 DWORDs (64B = 32 f16) insert 2 DWORDs (8B) -> LDS row
// stride 72B = 36 u16, matching the padded LDS layout used by the WMMA readers.
__device__ __forceinline__ void tdm_load_2d(const void* gsrc, void* lds_dst,
                                            unsigned tile_d0, unsigned tile_d1,
                                            unsigned tensor_d0, unsigned tensor_d1,
                                            unsigned stride_elems) {
    unsigned long long ga = (unsigned long long)(size_t)gsrc;
    unsigned lds_off = (unsigned)(size_t)lds_dst;
    u32x4 g0;
    g0[0] = 1u;                                            // count=1, user descriptor
    g0[1] = lds_off;                                       // lds_addr
    g0[2] = (unsigned)(ga & 0xffffffffu);                  // global_addr[31:0]
    g0[3] = (unsigned)((ga >> 32) & 0x1ffffffu) | (2u << 30); // addr[56:32] | type=2
    i32x8 g1;
    g1[0] = (int)((1u << 16) | (1u << 20) | (3u << 22) | (1u << 25));
    //            data_size=2B | pad_en | pad_interval=16DW | pad_amount=2DW
    g1[1] = (int)((tensor_d0 & 0xffffu) << 16);                                  // dim0[15:0]
    g1[2] = (int)(((tensor_d0 >> 16) & 0xffffu) | ((tensor_d1 & 0xffffu) << 16));// dim0[31:16]|dim1[15:0]
    g1[3] = (int)(((tensor_d1 >> 16) & 0xffffu) | ((tile_d0 & 0xffffu) << 16));  // dim1[31:16]|tile0
    g1[4] = (int)(tile_d1 & 0xffffu);                      // tile_dim1, tile_dim2=0
    g1[5] = (int)stride_elems;                             // tensor_dim0_stride[31:0]
    g1[6] = 0;
    g1[7] = 0;
    i32x4 z4 = {0, 0, 0, 0};
    i32x8 z8 = {0, 0, 0, 0, 0, 0, 0, 0};
    __builtin_amdgcn_tensor_load_to_lds(g0, g1, z4, z4, z8, 0);
}
#else
static constexpr int NBUF = 1;
#endif

// ---------------- workspace layout (bytes) ----------------
static constexpr size_t OFF_XP = 0;
static constexpr size_t SZ_XP  = (size_t)NB*NP*C2_*2;
static constexpr size_t OFF_QT = OFF_XP + SZ_XP;
static constexpr size_t SZ_QT  = (size_t)NB*NP*CQK_*2;
static constexpr size_t OFF_KT = OFF_QT + SZ_QT;
static constexpr size_t OFF_VB = OFF_KT + SZ_QT;
static constexpr size_t SZ_V   = (size_t)NB*C2_*NP*2;
static constexpr size_t OFF_VH = OFF_VB + SZ_V;
static constexpr size_t OFF_EN = OFF_VH + SZ_V;
static constexpr size_t SZ_EN  = (size_t)NB*NP*256*4;
static constexpr size_t OFF_AT = OFF_EN + SZ_EN;
static constexpr size_t SZ_AT  = (size_t)NB*NP*256*2;
static constexpr size_t OFF_YH = OFF_AT + SZ_AT;
static constexpr size_t OFF_YW = OFF_YH + SZ_V;
static constexpr size_t OFF_WQ = OFF_YW + SZ_V;
static constexpr size_t OFF_WK = OFF_WQ + (size_t)CQK_*C2_*2;
static constexpr size_t OFF_WV = OFF_WK + (size_t)CQK_*C2_*2;
static constexpr size_t OFF_WM = OFF_WV + (size_t)C2_*C2_*2;

// ---------------- K0: concat + transpose to pixel-major f16 ----------------
__global__ void k_pack_x(const float* __restrict__ x1, const float* __restrict__ x2,
                         h16* __restrict__ xp) {
    __shared__ float tile[32][33];
    int b  = blockIdx.z;
    int c0 = blockIdx.y * 32;
    int p0 = blockIdx.x * 32;
    int t  = threadIdx.x;
    for (int i = 0; i < 4; ++i) {
        int e  = t + i * 256;
        int cl = e >> 5, pl = e & 31;
        int c  = c0 + cl;
        const float* src = (c < CIN_) ? (x1 + ((size_t)b * CIN_ + c) * NP)
                                      : (x2 + ((size_t)b * CIN_ + (c - CIN_)) * NP);
        tile[cl][pl] = src[p0 + pl];
    }
    __syncthreads();
    for (int i = 0; i < 4; ++i) {
        int e  = t + i * 256;
        int pl = e >> 5, cl = e & 31;
        xp[((size_t)b * NP + p0 + pl) * C2_ + c0 + cl] = (h16)tile[cl][pl];
    }
}

// ---------------- K1: weights f32 -> f16 ----------------
__global__ void k_pack_w(const float* __restrict__ wq, const float* __restrict__ wk,
                         const float* __restrict__ wv, const float* __restrict__ wm,
                         h16* oq, h16* ok, h16* ov, h16* om) {
    int i = blockIdx.x * 256 + threadIdx.x;
    const int nqk = CQK_ * C2_;
    const int nvm = C2_ * C2_;
    if (i < nqk) { oq[i] = (h16)wq[i]; ok[i] = (h16)wk[i]; }
    if (i < nvm) { ov[i] = (h16)wv[i]; om[i] = (h16)wm[i]; }
}

// ---------------- K2: projection GEMM, MT n-tiles per wave ----------------
// MODE 0: transposed pixel-major f16 store (q,k).  MODE 1: dual store vb + vh.
template <int MODE, int MT>
__global__ void k_gemm_proj(const h16* __restrict__ Wmat, const float* __restrict__ bias,
                            const h16* __restrict__ xp, int M,
                            h16* __restrict__ out0, h16* __restrict__ out1) {
    constexpr int NWM = 8 / MT;                 // waves per m-tile
    constexpr int BUF = 128 * 36;
    __shared__ unsigned short bt[NBUF * BUF];   // [pixel][channel], 72B rows
    int b  = blockIdx.z;
    int n0 = blockIdx.x * 128;
    int m0 = blockIdx.y * (MT * 16);
    int t  = threadIdx.x;
    int wid = t >> 5, lane = t & 31, hl = lane >> 4, rw = lane & 15, kk = hl * 16;
    int m_t = wid / NWM, ng = wid % NWM;
    const h16* xpb = xp + (size_t)b * NP * C2_;
    v8f acc[MT];
#pragma unroll
    for (int i = 0; i < MT; ++i) acc[i] = (v8f){};
#if HAS_TDM
    if (wid == 0)
        tdm_load_2d(xpb + (size_t)n0 * C2_, bt, 32, 128, C2_, NP, C2_);
#endif
    for (int k0 = 0; k0 < C2_; k0 += 32) {
#if HAS_TDM
        int ph = (k0 >> 5) & 1;
        const unsigned short* btc = bt + ph * BUF;
        if (wid == 0) __builtin_amdgcn_s_wait_tensorcnt(0);
        __syncthreads();     // publish buffer ph; all reads of buffer ph^1 are done
        if (wid == 0 && k0 + 32 < C2_)
            tdm_load_2d(xpb + (size_t)n0 * C2_ + k0 + 32, bt + (ph ^ 1) * BUF,
                        32, 128, C2_, NP, C2_);
#else
        const unsigned short* btc = bt;
        {
            int pl = t & 127, ch = t >> 7;
            const h16* src = xpb + (size_t)(n0 + pl) * C2_ + k0 + ch * 16;
            if (k0 + 32 < C2_) __builtin_prefetch(src + 32, 0, 1);
            Frag o;
            o.q[0] = *(const uint4*)src;
            o.q[1] = *(const uint4*)(src + 8);
            uint2* dst = (uint2*)&bt[pl * 36 + ch * 16];
            dst[0] = o.d[0]; dst[1] = o.d[1]; dst[2] = o.d[2]; dst[3] = o.d[3];
        }
        __syncthreads();
#endif
        Frag af;
        const h16* arow = Wmat + (size_t)(m0 + m_t * 16 + rw) * C2_ + k0;
        af.q[0] = *(const uint4*)(arow + kk);
        af.q[1] = *(const uint4*)(arow + kk + 8);
#pragma unroll
        for (int tt = 0; tt < MT; ++tt) {
            Frag bf;
            const unsigned short* bp = &btc[((ng * MT + tt) * 16 + rw) * 36 + hl * 16];
            bf.d[0] = *(const uint2*)(bp + 0);  bf.d[1] = *(const uint2*)(bp + 4);
            bf.d[2] = *(const uint2*)(bp + 8);  bf.d[3] = *(const uint2*)(bp + 12);
            acc[tt] = __builtin_amdgcn_wmma_f32_16x16x32_f16(false, af.v, false, bf.v,
                                                             (short)0, acc[tt], false, false);
        }
#if !HAS_TDM
        __syncthreads();
#endif
    }
#pragma unroll
    for (int tt = 0; tt < MT; ++tt) {
        int n = n0 + (ng * MT + tt) * 16 + rw;
        if (MODE == 0) {
            Frag of;
            for (int r = 0; r < 8; ++r)
                of.h[r] = (h16)(acc[tt][r] + bias[m0 + m_t * 16 + hl * 8 + r]);
            *(uint4*)(out0 + ((size_t)b * NP + n) * M + m0 + m_t * 16 + hl * 8) = of.q[0];
        } else {
            for (int r = 0; r < 8; ++r) {
                int c = m0 + m_t * 16 + hl * 8 + r;
                h16 hv = (h16)(acc[tt][r] + bias[c]);
                out0[((size_t)b * C2_ + c) * NP + n] = hv;
                out1[((size_t)b * C2_ + c) * NP + (size_t)(n & (W_ - 1)) * H_ + (n >> 7)] = hv;
            }
        }
    }
}

// ---------------- K3: energy GEMM, K = 32, 2 j-tiles per wave ----------------
// RM 0: column attention (line = x, writes energy[...][0:128))
// RM 1: row attention    (line = y, writes energy[...][128:256))
template <int RM>
__global__ void k_energy(const h16* __restrict__ qt, const h16* __restrict__ kt,
                         float* __restrict__ energy) {
    int b = blockIdx.z;
    int L = blockIdx.y;
    int i0 = blockIdx.x * 32;
    int t = threadIdx.x;
    int wid = t >> 5, lane = t & 31, hl = lane >> 4, rw = lane & 15, kk = hl * 16;
    int it = wid >> 2, jg = wid & 3;
    int i = i0 + it * 16 + rw;
    size_t pA = (RM == 0) ? ((size_t)i * W_ + L) : ((size_t)L * W_ + i);
    Frag af;
    const h16* ab = qt + ((size_t)b * NP + pA) * CQK_;
    af.q[0] = *(const uint4*)(ab + kk);
    af.q[1] = *(const uint4*)(ab + kk + 8);
    v8f acc[2];
#pragma unroll
    for (int jt = 0; jt < 2; ++jt) {
        int j = (jg * 2 + jt) * 16 + rw;
        size_t pB = (RM == 0) ? ((size_t)j * W_ + L) : ((size_t)L * W_ + j);
        Frag bf;
        const h16* bb = kt + ((size_t)b * NP + pB) * CQK_ + hl * 16;
        bf.q[0] = *(const uint4*)(bb);
        bf.q[1] = *(const uint4*)(bb + 8);
        acc[jt] = (v8f){};
        acc[jt] = __builtin_amdgcn_wmma_f32_16x16x32_f16(false, af.v, false, bf.v,
                                                         (short)0, acc[jt], false, false);
    }
#pragma unroll
    for (int jt = 0; jt < 2; ++jt) {
        for (int r = 0; r < 8; ++r) {
            int ii = i0 + it * 16 + hl * 8 + r;
            int jj = (jg * 2 + jt) * 16 + (lane & 15);
            size_t pix = (RM == 0) ? ((size_t)ii * W_ + L) : ((size_t)L * W_ + ii);
            energy[((size_t)b * NP + pix) * 256 + (RM == 0 ? 0 : 128) + jj] = acc[jt][r];
        }
    }
}

// ---------------- K4: softmax over 256 (one wave per pixel) ----------------
__global__ void k_softmax(const float* __restrict__ energy, h16* __restrict__ att) {
    int t = threadIdx.x;
    int wid = t >> 5, lane = t & 31;
    size_t pixel = (size_t)blockIdx.x * 8 + wid;
    int p = (int)(pixel & (NP - 1));
    int yy = p >> 7;
    const float4* src = (const float4*)(energy + pixel * 256 + lane * 8);
    float4 v0 = src[0], v1 = src[1];
    float vals[8] = {v0.x, v0.y, v0.z, v0.w, v1.x, v1.y, v1.z, v1.w};
    float mx = -3.0e38f;
    for (int e = 0; e < 8; ++e) {
        int idx = lane * 8 + e;
        if (idx < 128 && idx == yy) vals[e] = -3.0e38f;   // diagonal mask of energy_H
        mx = fmaxf(mx, vals[e]);
    }
    for (int m = 16; m >= 1; m >>= 1) mx = fmaxf(mx, __shfl_xor(mx, m, 32));
    float sum = 0.f;
    for (int e = 0; e < 8; ++e) { vals[e] = __expf(vals[e] - mx); sum += vals[e]; }
    for (int m = 16; m >= 1; m >>= 1) sum += __shfl_xor(sum, m, 32);
    float inv = 1.0f / sum;
    Frag of;
    for (int e = 0; e < 8; ++e) of.h[e] = (h16)(vals[e] * inv);
    *(uint4*)(att + pixel * 256 + lane * 8) = of.q[0];
}

// ---------------- K5: apply attention: 4 c-tiles per wave ----------------
// RM 0: column attention (vsrc = vh, att offset 0); RM 1: row (vsrc = vb, offset 128)
template <int RM>
__global__ void k_outattn(const h16* __restrict__ att, const h16* __restrict__ vsrc,
                          h16* __restrict__ yout) {
    constexpr int BUF = 64 * 36;
    __shared__ unsigned short vt[NBUF * BUF];   // [c_local][j_local], 72B rows
    int b  = blockIdx.z;
    int L  = blockIdx.y;
    int c0 = blockIdx.x * 64;
    int t  = threadIdx.x;
    int wid = t >> 5, lane = t & 31, hl = lane >> 4, rw = lane & 15, kk = hl * 16;
    int i = wid * 16 + rw;
    const h16* vbase = vsrc + ((size_t)b * C2_ + c0) * NP + (size_t)L * W_;
    v8f acc[4];
#pragma unroll
    for (int q = 0; q < 4; ++q) acc[q] = (v8f){};
#if HAS_TDM
    if (wid == 0)
        tdm_load_2d(vbase, vt, 32, 64, NP, C2_, NP);
#endif
    for (int k0 = 0; k0 < 128; k0 += 32) {
#if HAS_TDM
        int ph = (k0 >> 5) & 1;
        const unsigned short* vtc = vt + ph * BUF;
        if (wid == 0) __builtin_amdgcn_s_wait_tensorcnt(0);
        __syncthreads();
        if (wid == 0 && k0 + 32 < 128)
            tdm_load_2d(vbase + k0 + 32, vt + (ph ^ 1) * BUF, 32, 64, NP, C2_, NP);
#else
        const unsigned short* vtc = vt;
        {
            int cl = t >> 2, j8 = (t & 3) * 8;
            const uint4* s = (const uint4*)(vbase + (size_t)cl * NP + k0 + j8);
            Frag o; o.q[0] = *s;
            uint2* dst = (uint2*)&vt[cl * 36 + j8];
            dst[0] = o.d[0]; dst[1] = o.d[1];
        }
        __syncthreads();
#endif
        size_t pixA = (RM == 0) ? ((size_t)i * W_ + L) : ((size_t)L * W_ + i);
        const h16* abase = att + ((size_t)b * NP + pixA) * 256 + (RM == 0 ? 0 : 128) + k0;
        Frag af;
        af.q[0] = *(const uint4*)(abase + kk);
        af.q[1] = *(const uint4*)(abase + kk + 8);
#pragma unroll
        for (int ct = 0; ct < 4; ++ct) {
            Frag bf;
            const unsigned short* bp = &vtc[(ct * 16 + rw) * 36 + hl * 16];
            bf.d[0] = *(const uint2*)(bp + 0);  bf.d[1] = *(const uint2*)(bp + 4);
            bf.d[2] = *(const uint2*)(bp + 8);  bf.d[3] = *(const uint2*)(bp + 12);
            acc[ct] = __builtin_amdgcn_wmma_f32_16x16x32_f16(false, af.v, false, bf.v,
                                                             (short)0, acc[ct], false, false);
        }
#if !HAS_TDM
        __syncthreads();
#endif
    }
#pragma unroll
    for (int ct = 0; ct < 4; ++ct) {
        for (int r = 0; r < 8; ++r) {
            int ii = wid * 16 + hl * 8 + r;
            int c  = c0 + ct * 16 + (lane & 15);
            size_t pix = (RM == 0) ? ((size_t)ii * W_ + L) : ((size_t)L * W_ + ii);
            yout[((size_t)b * NP + pix) * C2_ + c] = (h16)acc[ct][r];
        }
    }
}

// ---------------- K6: final conv with fused input build, 4 tiles per wave ----------------
__global__ void k_final(const h16* __restrict__ Wm, const float* __restrict__ bm,
                        const h16* __restrict__ xp, const h16* __restrict__ yH,
                        const h16* __restrict__ yW, const float* __restrict__ gptr,
                        float* __restrict__ out) {
    __shared__ unsigned short bt[128 * 36];
    int b  = blockIdx.z;
    int n0 = blockIdx.x * 128;
    int m0 = blockIdx.y * 64;
    int t  = threadIdx.x;
    int wid = t >> 5, lane = t & 31, hl = lane >> 4, rw = lane & 15, kk = hl * 16;
    int m_t = wid >> 1, ng = wid & 1;
    float g = gptr[0];
    v8f acc[4];
#pragma unroll
    for (int q = 0; q < 4; ++q) acc[q] = (v8f){};
    for (int k0 = 0; k0 < C2_; k0 += 32) {
        {   // build fused input tile in LDS: gamma*(yH+yW) + x
            int pl = t & 127, ch = t >> 7;
            size_t base = ((size_t)b * NP + n0 + pl) * C2_ + k0 + ch * 16;
            if (k0 + 32 < C2_) __builtin_prefetch(xp + base + 32, 0, 1);
            Frag xh, hh, wh, o;
            xh.q[0] = *(const uint4*)(xp + base); xh.q[1] = *(const uint4*)(xp + base + 8);
            hh.q[0] = *(const uint4*)(yH + base); hh.q[1] = *(const uint4*)(yH + base + 8);
            wh.q[0] = *(const uint4*)(yW + base); wh.q[1] = *(const uint4*)(yW + base + 8);
            for (int e = 0; e < 16; ++e)
                o.h[e] = (h16)(g * ((float)hh.h[e] + (float)wh.h[e]) + (float)xh.h[e]);
            uint2* dst = (uint2*)&bt[pl * 36 + ch * 16];
            dst[0] = o.d[0]; dst[1] = o.d[1]; dst[2] = o.d[2]; dst[3] = o.d[3];
        }
        __syncthreads();
        Frag af;
        const h16* arow = Wm + (size_t)(m0 + m_t * 16 + rw) * C2_ + k0;
        af.q[0] = *(const uint4*)(arow + kk);
        af.q[1] = *(const uint4*)(arow + kk + 8);
#pragma unroll
        for (int tt = 0; tt < 4; ++tt) {
            Frag bf;
            const unsigned short* bp = &bt[((ng * 4 + tt) * 16 + rw) * 36 + hl * 16];
            bf.d[0] = *(const uint2*)(bp + 0);  bf.d[1] = *(const uint2*)(bp + 4);
            bf.d[2] = *(const uint2*)(bp + 8);  bf.d[3] = *(const uint2*)(bp + 12);
            acc[tt] = __builtin_amdgcn_wmma_f32_16x16x32_f16(false, af.v, false, bf.v,
                                                             (short)0, acc[tt], false, false);
        }
        __syncthreads();
    }
#pragma unroll
    for (int tt = 0; tt < 4; ++tt) {
        int n = n0 + (ng * 4 + tt) * 16 + (lane & 15);
        for (int r = 0; r < 8; ++r) {
            int o_ch = m0 + m_t * 16 + hl * 8 + r;
            float vv = acc[tt][r] + bm[o_ch];
            int gsel = o_ch >> 8;   // out.reshape(b,2,256,h,w).transpose(1,0,2,3,4)
            int cp   = o_ch & 255;
            out[(((size_t)gsel * NB + b) * 256 + cp) * NP + n] = vv;
        }
    }
}

// ---------------- host ----------------
extern "C" void kernel_launch(void* const* d_in, const int* in_sizes, int n_in,
                              void* d_out, int out_size, void* d_ws, size_t ws_size,
                              hipStream_t stream) {
    (void)in_sizes; (void)n_in; (void)out_size; (void)ws_size;
    const float* x1 = (const float*)d_in[0];
    const float* x2 = (const float*)d_in[1];
    const float* Wq = (const float*)d_in[2];
    const float* bq = (const float*)d_in[3];
    const float* Wk = (const float*)d_in[4];
    const float* bk = (const float*)d_in[5];
    const float* Wv = (const float*)d_in[6];
    const float* bv = (const float*)d_in[7];
    const float* Wm = (const float*)d_in[8];
    const float* bm = (const float*)d_in[9];
    const float* gm = (const float*)d_in[10];
    float* out = (float*)d_out;
    char* ws = (char*)d_ws;

    h16* xp  = (h16*)(ws + OFF_XP);
    h16* qt  = (h16*)(ws + OFF_QT);
    h16* kt  = (h16*)(ws + OFF_KT);
    h16* vb  = (h16*)(ws + OFF_VB);
    h16* vh  = (h16*)(ws + OFF_VH);
    float* en = (float*)(ws + OFF_EN);
    h16* at  = (h16*)(ws + OFF_AT);
    h16* yH  = (h16*)(ws + OFF_YH);
    h16* yW  = (h16*)(ws + OFF_YW);
    h16* wqh = (h16*)(ws + OFF_WQ);
    h16* wkh = (h16*)(ws + OFF_WK);
    h16* wvh = (h16*)(ws + OFF_WV);
    h16* wmh = (h16*)(ws + OFF_WM);

    dim3 thr(256);
    k_pack_x<<<dim3(NP / 32, C2_ / 32, NB), thr, 0, stream>>>(x1, x2, xp);
    k_pack_w<<<dim3((C2_ * C2_) / 256), thr, 0, stream>>>(Wq, Wk, Wv, Wm, wqh, wkh, wvh, wmh);
    k_gemm_proj<0, 2><<<dim3(NP / 128, 1, NB), thr, 0, stream>>>(wqh, bq, xp, CQK_, qt, nullptr);
    k_gemm_proj<0, 2><<<dim3(NP / 128, 1, NB), thr, 0, stream>>>(wkh, bk, xp, CQK_, kt, nullptr);
    k_gemm_proj<1, 4><<<dim3(NP / 128, C2_ / 64, NB), thr, 0, stream>>>(wvh, bv, xp, C2_, vb, vh);
    k_energy<0><<<dim3(H_ / 32, W_, NB), thr, 0, stream>>>(qt, kt, en);
    k_energy<1><<<dim3(W_ / 32, H_, NB), thr, 0, stream>>>(qt, kt, en);
    k_softmax<<<dim3(NB * NP / 8), thr, 0, stream>>>(en, at);
    k_outattn<0><<<dim3(C2_ / 64, W_, NB), thr, 0, stream>>>(at, vh, yH);
    k_outattn<1><<<dim3(C2_ / 64, H_, NB), thr, 0, stream>>>(at, vb, yW);
    k_final<<<dim3(NP / 128, C2_ / 64, NB), thr, 0, stream>>>(wmh, bm, xp, yH, yW, gm, out);
}
